// Block_88252987998571
// MI455X (gfx1250) — matrix operand order repaired and checked by reading, not compiled, not run on hardware
//
#include <hip/hip_runtime.h>
#include <hip/hip_bf16.h>

typedef _Float16 v8h  __attribute__((ext_vector_type(8)));
typedef _Float16 v16h __attribute__((ext_vector_type(16)));
typedef float    v8f  __attribute__((ext_vector_type(8)));
typedef unsigned int v4u __attribute__((ext_vector_type(4)));
typedef int      v8i  __attribute__((ext_vector_type(8)));
typedef int      v4i  __attribute__((ext_vector_type(4)));

constexpr int kB  = 128;
constexpr int kT  = 256;
constexpr int kC  = 384;
constexpr int kH  = 6;
constexpr int kHD = 64;
constexpr int kBT = kB * kT;      // 32768 rows
constexpr int kF  = 4 * kC;       // 1536
constexpr float kEps   = 1e-5f;
constexpr float kScale = 0.125f;  // HD^-0.5

// Attention LDS layout (bytes): K tile [256][72]h, V^T tile [64][264]h,
// P staging [8 waves][16][264]h.
constexpr unsigned kLdsK = 0;
constexpr unsigned kLdsV = 256 * 72 * 2;                  // 36864
constexpr unsigned kLdsP = kLdsV + 64 * 264 * 2;          // 70656
constexpr unsigned kLdsTotal = kLdsP + 8 * 16 * 264 * 2;  // 138240

// ---------------------------------------------------------------------------
// WMMA helpers (CDNA5 V_WMMA_F32_16X16X32_F16, wave32)
// ---------------------------------------------------------------------------
__device__ __forceinline__ v8f wmma_f16(v16h a, v16h b, v8f c) {
  return __builtin_amdgcn_wmma_f32_16x16x32_f16(false, a, false, b, (short)0, c,
                                                false, false);
}

__device__ __forceinline__ v16h cat8(v8h lo, v8h hi) {
  return __builtin_shufflevector(lo, hi, 0, 1, 2, 3, 4, 5, 6, 7, 8, 9, 10, 11,
                                 12, 13, 14, 15);
}

// A fragment (16x32 f16): lane<16 holds row=lane, K {0..7,16..23}; lane>=16
// holds row=lane-16, K {8..15,24..31}.
__device__ __forceinline__ v16h frag_a(const _Float16* base, int stride,
                                       int lane, int kbase) {
  const _Float16* p =
      base + (size_t)(lane & 15) * stride + kbase + ((lane >> 4) << 3);
  return cat8(*(const v8h*)p, *(const v8h*)(p + 16));
}

// B fragment (32x16 f16): lane holds col N=lane&15, K {0..7,16..23} (lane<16)
// or K {8..15,24..31} (lane>=16), read from an N-major "B-transposed" tile.
__device__ __forceinline__ v16h frag_b(const _Float16* base, int stride,
                                       int lane, int kbase) {
  const _Float16* p =
      base + (size_t)(lane & 15) * stride + kbase + ((lane >> 4) << 3);
  return cat8(*(const v8h*)p, *(const v8h*)(p + 16));
}

// ---------------------------------------------------------------------------
// CDNA5 async global->LDS DMA (ASYNCcnt-tracked).
// ---------------------------------------------------------------------------
__device__ __forceinline__ void async_copy_b128(const _Float16* lds_dst,
                                                const _Float16* gsrc) {
  unsigned int loff = (unsigned int)(size_t)lds_dst;
  asm volatile("global_load_async_to_lds_b128 %0, %1, off" ::"v"(loff),
               "v"(gsrc)
               : "memory");
}

__device__ __forceinline__ void wait_async0() {
  asm volatile("s_wait_asynccnt 0x0" ::: "memory");
}

// ---------------------------------------------------------------------------
// CDNA5 Tensor Data Mover: one-descriptor 2D tile load global->LDS with
// row padding done by the TDM (D# groups per CDNA5 ISA).
// dim0/dim1 in elements (2B), tile == whole tensor, pad codes per ISA.
// This toolchain's builtin takes 6 args: (g0 u32x4, g1 i32x8, g2 i32x4,
// g3 i32x4, extra i32x8, cpol i32).
// ---------------------------------------------------------------------------
#if __has_builtin(__builtin_amdgcn_tensor_load_to_lds)
#define HAVE_TDM 1
__device__ __forceinline__ void tdm_load_2d(unsigned lds_off, const void* gptr,
                                            unsigned dim0, unsigned dim1,
                                            unsigned stride0, unsigned pad_int,
                                            unsigned pad_amt) {
  unsigned long long ga = (unsigned long long)(size_t)gptr;
  v4u g0;
  g0[0] = 1u;                                    // count=1, user descriptor
  g0[1] = lds_off;                               // lds_addr (bytes)
  g0[2] = (unsigned)(ga & 0xFFFFFFFFu);          // global_addr[31:0]
  g0[3] = (unsigned)((ga >> 32) & 0x01FFFFFFu)   // global_addr[56:32]
          | (2u << 30);                          // type=2 ("image")
  v8i g1;
  g1[0] = (int)((1u << 16)                       // data_size = 2B
                | (1u << 20)                     // pad_enable
                | (pad_int << 22)                // pad_interval code
                | (pad_amt << 25));              // pad_amount code
  g1[1] = (int)((dim0 & 0xFFFFu) << 16);         // tensor_dim0[15:0]
  g1[2] = (int)(((dim0 >> 16) & 0xFFFFu) | ((dim1 & 0xFFFFu) << 16));
  g1[3] = (int)(((dim1 >> 16) & 0xFFFFu) | ((dim0 & 0xFFFFu) << 16));  // tile_dim0
  g1[4] = (int)(dim1 & 0xFFFFu);                 // tile_dim1 (tile_dim2=0)
  g1[5] = (int)stride0;                          // tensor_dim0_stride[31:0]
  g1[6] = 0;
  g1[7] = 0;
  v4i z4 = {0, 0, 0, 0};
  v8i z8 = {0, 0, 0, 0, 0, 0, 0, 0};
  __builtin_amdgcn_tensor_load_to_lds(g0, g1, z4, z4, z8, 0);
}
#else
#define HAVE_TDM 0
#endif

// ---------------------------------------------------------------------------
// Weight prep: dst[g][n][k] = (fp16) src[g][k][n]
// ---------------------------------------------------------------------------
__global__ void convert_transpose(const float* __restrict__ src,
                                  _Float16* __restrict__ dst, int K, int N) {
  size_t plane = (size_t)K * N;
  const float* s = src + (size_t)blockIdx.z * plane;
  _Float16* d = dst + (size_t)blockIdx.z * plane;
  int idx = blockIdx.x * 256 + threadIdx.x;
  if (idx < (int)plane) {
    int n = idx / K;
    int k = idx - n * K;
    d[idx] = (_Float16)s[(size_t)k * N + n];
  }
}

// ---------------------------------------------------------------------------
// LayerNorm (fp32 stats) -> fp16 rows.  One wave per row of C=384.
// ---------------------------------------------------------------------------
__global__ __launch_bounds__(256) void ln_k(const float* __restrict__ x,
                                            const float* __restrict__ g,
                                            const float* __restrict__ beta,
                                            _Float16* __restrict__ out) {
  int wave = threadIdx.x >> 5, lane = threadIdx.x & 31;
  int row = blockIdx.x * 8 + wave;
  const float* xr = x + (size_t)row * kC;
  float vals[12], s = 0.f, s2 = 0.f;
#pragma unroll
  for (int i = 0; i < 12; i++) {
    float v = xr[lane + 32 * i];
    vals[i] = v;
    s += v;
    s2 += v * v;
  }
#pragma unroll
  for (int off = 16; off; off >>= 1) {
    s += __shfl_xor(s, off, 32);
    s2 += __shfl_xor(s2, off, 32);
  }
  float mu = s * (1.0f / kC);
  float var = s2 * (1.0f / kC) - mu * mu;
  float r = rsqrtf(var + kEps);
  _Float16* o = out + (size_t)row * kC;
#pragma unroll
  for (int i = 0; i < 12; i++) {
    int c = lane + 32 * i;
    o[c] = (_Float16)((vals[i] - mu) * r * g[c] + beta[c]);
  }
}

// ---------------------------------------------------------------------------
// Generic WMMA GEMM:  D[m][n] = sum_k A[m][k] * Bt[n][k]
// Block tile 128(M) x 64(N), 8 waves of 32x32, K-step 32, double-buffered
// LDS filled by async global->LDS DMA overlapping the WMMAs.
// ---------------------------------------------------------------------------
template <int MODE>
__global__ __launch_bounds__(256) void gemm_k(
    const _Float16* __restrict__ A, const _Float16* __restrict__ Bt, int K,
    int N, float* __restrict__ fout, const float* __restrict__ res,
    const float* __restrict__ bias, _Float16* __restrict__ hout,
    _Float16* __restrict__ qp, _Float16* __restrict__ kp,
    _Float16* __restrict__ vp) {
  __shared__ _Float16 sA[2][128 * 40];
  __shared__ _Float16 sB[2][64 * 40];
  const int tid = threadIdx.x;
  const int lane = tid & 31, wave = tid >> 5;
  const int wm = wave & 3, wn = wave >> 2;
  const int m0 = blockIdx.x * 128, n0 = blockIdx.y * 64;

  const int rowA0 = tid >> 1, offA0 = (tid & 1) << 4;
  const int rowB = tid >> 2, offB = (tid & 3) << 3;

  auto stage = [&](int buf, int k0) {
    const _Float16* ga = A + (size_t)(m0 + rowA0) * K + k0 + offA0;
    _Float16* la = &sA[buf][rowA0 * 40 + offA0];
    async_copy_b128(la, ga);
    async_copy_b128(la + 8, ga + 8);
    async_copy_b128(&sB[buf][rowB * 40 + offB],
                    Bt + (size_t)(n0 + rowB) * K + k0 + offB);
  };

  v8f zero = {};
  v8f acc[2][2];
#pragma unroll
  for (int mi = 0; mi < 2; mi++)
#pragma unroll
    for (int ni = 0; ni < 2; ni++) acc[mi][ni] = zero;

  const int nk = K >> 5;
  stage(0, 0);
  for (int i = 0; i < nk; i++) {
    const int cur = i & 1;
    wait_async0();
    __syncthreads();
    if (i + 1 < nk) stage(cur ^ 1, (i + 1) << 5);

    v16h af[2], bf[2];
#pragma unroll
    for (int mi = 0; mi < 2; mi++)
      af[mi] = frag_a(&sA[cur][(wm * 32 + mi * 16) * 40], 40, lane, 0);
#pragma unroll
    for (int ni = 0; ni < 2; ni++)
      bf[ni] = frag_b(&sB[cur][(wn * 32 + ni * 16) * 40], 40, lane, 0);
#pragma unroll
    for (int mi = 0; mi < 2; mi++)
#pragma unroll
      for (int ni = 0; ni < 2; ni++)
        acc[mi][ni] = wmma_f16(af[mi], bf[ni], acc[mi][ni]);
  }

  const int hl = lane >> 4, nl = lane & 15;
#pragma unroll
  for (int mi = 0; mi < 2; mi++) {
#pragma unroll
    for (int ni = 0; ni < 2; ni++) {
      v8f c = acc[mi][ni];
      int n = n0 + wn * 32 + ni * 16 + nl;
      int mbase = m0 + wm * 32 + mi * 16 + hl * 8;
      if constexpr (MODE == 0) {
        int w = n / kC;  // 0=q 1=k 2=v
        int r = n - w * kC;
        int hh = r >> 6, d = r & 63;
        _Float16* dst = (w == 0) ? qp : ((w == 1) ? kp : vp);
#pragma unroll
        for (int j = 0; j < 8; j++) {
          int m = mbase + j;
          int bb = m >> 8, t = m & 255;
          float val = c[j];
          if (w == 0) val *= kScale;
          size_t idx;
          if (w == 2)
            idx = (((size_t)bb * kH + hh) * kHD + d) * kT + t;  // v^T [d][t]
          else
            idx = (((size_t)bb * kH + hh) * kT + t) * kHD + d;
          dst[idx] = (_Float16)val;
        }
      } else if constexpr (MODE == 1) {
        float bv = bias[n];
#pragma unroll
        for (int j = 0; j < 8; j++) {
          int m = mbase + j;
          float val = c[j] + bv;
          hout[(size_t)m * N + n] = (_Float16)(val > 0.f ? val : 0.f);
        }
      } else {
        float bv = bias[n];
#pragma unroll
        for (int j = 0; j < 8; j++) {
          int m = mbase + j;
          fout[(size_t)m * N + n] = c[j] + bv + res[(size_t)m * N + n];
        }
      }
    }
  }
}

// ---------------------------------------------------------------------------
// Causal attention, one (b,h) x 128 query rows per block; 8 waves x 16 rows.
// K and V^T tiles are DMA'd into LDS once per block by the Tensor Data Mover
// (TDM pads each row so the LDS strides are bank-conflict friendly);
// scores live in 16 v8f accumulators; softmax in registers; P staged via LDS.
// ---------------------------------------------------------------------------
__global__ __launch_bounds__(256) void attn_k(const _Float16* __restrict__ q,
                                              const _Float16* __restrict__ kk,
                                              const _Float16* __restrict__ vT,
                                              _Float16* __restrict__ attn_h) {
  extern __shared__ char smem_raw[];
  _Float16* kL = (_Float16*)(smem_raw + kLdsK);  // [256][72]
  _Float16* vL = (_Float16*)(smem_raw + kLdsV);  // [64][264]
  _Float16* pL = (_Float16*)(smem_raw + kLdsP);  // [8][16][264]

  const int tid = threadIdx.x, wave = tid >> 5, lane = tid & 31;
  const int hl = lane >> 4, nl = lane & 15;
  const int bh = blockIdx.x;
  const int b = bh / kH, h = bh - b * kH;
  const int t0 = blockIdx.y * 128;
  const int tR = t0 + wave * 16;

  const _Float16* kg = kk + (size_t)bh * kT * kHD;  // [256][64]
  const _Float16* vg = vT + (size_t)bh * kHD * kT;  // [64][256]

#if HAVE_TDM
  if (wave == 0) {
    // K: rows of 64 halfs = 32 DWORDs -> pad_interval code 4 (32 DW),
    //    pad_amount code 3 (4 DW = 8 halfs) -> LDS stride 72 halfs.
    tdm_load_2d((unsigned)(size_t)kL, kg, kHD, kT, kHD, 4u, 3u);
    // V^T: rows of 256 halfs = 128 DWORDs -> code 6 (128 DW), amount code 3
    //    -> LDS stride 264 halfs.
    tdm_load_2d((unsigned)(size_t)vL, vg, kT, kHD, kT, 6u, 3u);
    __builtin_amdgcn_s_wait_tensorcnt(0);
  }
#else
  // Fallback: cooperative 16B copies.
  for (int c = tid; c < 2048; c += 256) {  // K tile: 2048 chunks of 8 halfs
    int r = c >> 3, o = (c & 7) << 3;
    *(uint4*)&kL[r * 72 + o] = *(const uint4*)(kg + (size_t)r * kHD + o);
  }
  for (int c = tid; c < 2048; c += 256) {  // V^T tile
    int r = c >> 5, o = (c & 31) << 3;
    *(uint4*)&vL[r * 264 + o] = *(const uint4*)(vg + (size_t)r * kT + o);
  }
#endif
  __syncthreads();

  const _Float16* qg = q + ((size_t)bh * kT + tR) * kHD;
  v16h aq0 = frag_a(qg, kHD, lane, 0);
  v16h aq1 = frag_a(qg, kHD, lane, 32);

  v8f s[16];
#pragma unroll
  for (int nb = 0; nb < 16; nb++) {
    const _Float16* kbp = kL + (nb * 16) * 72;
    v8f a = {};
    a = wmma_f16(aq0, frag_b(kbp, 72, lane, 0), a);
    a = wmma_f16(aq1, frag_b(kbp, 72, lane, 32), a);
    s[nb] = a;
  }

#pragma unroll
  for (int j = 0; j < 8; j++) {
    int tq = tR + j + 8 * hl;
    float m = -1e30f;
#pragma unroll
    for (int nb = 0; nb < 16; nb++) {
      int sc = nb * 16 + nl;
      float v = (sc <= tq) ? s[nb][j] : -1e30f;
      s[nb][j] = v;
      m = fmaxf(m, v);
    }
#pragma unroll
    for (int off = 8; off; off >>= 1) m = fmaxf(m, __shfl_xor(m, off, 16));
    float sum = 0.f;
#pragma unroll
    for (int nb = 0; nb < 16; nb++) {
      float e = __expf(s[nb][j] - m);
      s[nb][j] = e;
      sum += e;
    }
#pragma unroll
    for (int off = 8; off; off >>= 1) sum += __shfl_xor(sum, off, 16);
    float inv = 1.0f / sum;
#pragma unroll
    for (int nb = 0; nb < 16; nb++) s[nb][j] *= inv;
  }

  _Float16* Pw = pL + (size_t)wave * 16 * 264;
#pragma unroll
  for (int j = 0; j < 8; j++) {
    int r = j + 8 * hl;
#pragma unroll
    for (int nb = 0; nb < 16; nb++)
      Pw[r * 264 + nb * 16 + nl] = (_Float16)s[nb][j];
  }
  __syncthreads();

#pragma unroll
  for (int db = 0; db < 4; db++) {
    v8f o = {};
#pragma unroll
    for (int ks = 0; ks < 8; ks++) {
      v16h ap = frag_a(Pw, 264, lane, ks * 32);
      v16h bv = frag_b(vL + (size_t)(db * 16) * 264, 264, lane, ks * 32);
      o = wmma_f16(ap, bv, o);
    }
#pragma unroll
    for (int j = 0; j < 8; j++) {
      int t = tR + j + 8 * hl;
      attn_h[((size_t)b * kT + t) * kC + h * kHD + db * 16 + nl] =
          (_Float16)o[j];
    }
  }
}

// ---------------------------------------------------------------------------
// Host orchestration
// ---------------------------------------------------------------------------
static inline size_t ws_alloc(size_t& off, size_t bytes) {
  size_t r = off;
  off = (off + bytes + 255) & ~(size_t)255;
  return r;
}

extern "C" void kernel_launch(void* const* d_in, const int* in_sizes, int n_in,
                              void* d_out, int out_size, void* d_ws,
                              size_t ws_size, hipStream_t stream) {
  const float* x     = (const float*)d_in[0];
  const float* ln1_g = (const float*)d_in[1];
  const float* ln1_b = (const float*)d_in[2];
  const float* Wq    = (const float*)d_in[3];
  const float* Wk    = (const float*)d_in[4];
  const float* Wv    = (const float*)d_in[5];
  const float* Wo    = (const float*)d_in[6];
  const float* bo    = (const float*)d_in[7];
  const float* ln2_g = (const float*)d_in[8];
  const float* ln2_b = (const float*)d_in[9];
  const float* W1    = (const float*)d_in[10];
  const float* b1    = (const float*)d_in[11];
  const float* W2    = (const float*)d_in[12];
  const float* b2    = (const float*)d_in[13];
  float* out = (float*)d_out;

  char* ws = (char*)d_ws;
  size_t off = 0;
  _Float16* x1h   = (_Float16*)(ws + ws_alloc(off, (size_t)kBT * kC * 2));
  _Float16* qb    = (_Float16*)(ws + ws_alloc(off, (size_t)kB * kH * kT * kHD * 2));
  _Float16* kb    = (_Float16*)(ws + ws_alloc(off, (size_t)kB * kH * kT * kHD * 2));
  _Float16* vTb   = (_Float16*)(ws + ws_alloc(off, (size_t)kB * kH * kT * kHD * 2));
  _Float16* atth  = (_Float16*)(ws + ws_alloc(off, (size_t)kBT * kC * 2));
  float*    ybuf  = (float*)   (ws + ws_alloc(off, (size_t)kBT * kC * 4));
  _Float16* x2h   = (_Float16*)(ws + ws_alloc(off, (size_t)kBT * kC * 2));
  _Float16* hbuf  = (_Float16*)(ws + ws_alloc(off, (size_t)kBT * kF * 2));
  _Float16* WqkvT = (_Float16*)(ws + ws_alloc(off, (size_t)3 * kC * kC * 2));
  _Float16* WoT   = (_Float16*)(ws + ws_alloc(off, (size_t)kC * kC * 2));
  _Float16* W1T   = (_Float16*)(ws + ws_alloc(off, (size_t)kC * kF * 2));
  _Float16* W2T   = (_Float16*)(ws + ws_alloc(off, (size_t)kC * kF * 2));

  // 0) weights -> fp16 transposed [N][K]
  convert_transpose<<<dim3((kC * kHD + 255) / 256, 1, kH), 256, 0, stream>>>(
      Wq, WqkvT + 0 * (size_t)kC * kC, kC, kHD);
  convert_transpose<<<dim3((kC * kHD + 255) / 256, 1, kH), 256, 0, stream>>>(
      Wk, WqkvT + 1 * (size_t)kC * kC, kC, kHD);
  convert_transpose<<<dim3((kC * kHD + 255) / 256, 1, kH), 256, 0, stream>>>(
      Wv, WqkvT + 2 * (size_t)kC * kC, kC, kHD);
  convert_transpose<<<dim3((kC * kC + 255) / 256, 1, 1), 256, 0, stream>>>(
      Wo, WoT, kC, kC);
  convert_transpose<<<dim3((kC * kF + 255) / 256, 1, 1), 256, 0, stream>>>(
      W1, W1T, kC, kF);
  convert_transpose<<<dim3((kC * kF + 255) / 256, 1, 1), 256, 0, stream>>>(
      W2, W2T, kF, kC);

  // 1) LN1
  ln_k<<<kBT / 8, 256, 0, stream>>>(x, ln1_g, ln1_b, x1h);

  // 2) fused QKV projection (M=32768, N=1152, K=384)
  gemm_k<0><<<dim3(kBT / 128, (3 * kC) / 64), 256, 0, stream>>>(
      x1h, WqkvT, kC, 3 * kC, nullptr, nullptr, nullptr, nullptr, qb, kb, vTb);

  // 3) attention (TDM-staged K/V^T tiles)
  {
    (void)hipFuncSetAttribute((const void*)attn_k,
                              hipFuncAttributeMaxDynamicSharedMemorySize,
                              (int)kLdsTotal);
    attn_k<<<dim3(kB * kH, kT / 128), 256, kLdsTotal, stream>>>(qb, kb, vTb,
                                                                atth);
  }

  // 4) out-projection + residual: y = x + attn @ Wo + bo
  gemm_k<2><<<dim3(kBT / 128, kC / 64), 256, 0, stream>>>(
      atth, WoT, kC, kC, ybuf, x, bo, nullptr, nullptr, nullptr, nullptr);

  // 5) LN2
  ln_k<<<kBT / 8, 256, 0, stream>>>(ybuf, ln2_g, ln2_b, x2h);

  // 6) FFN1: h = relu(x2 @ W1 + b1)
  gemm_k<1><<<dim3(kBT / 128, kF / 64), 256, 0, stream>>>(
      x2h, W1T, kC, kF, nullptr, nullptr, b1, hbuf, nullptr, nullptr, nullptr);

  // 7) FFN2 + residual: out = y + h @ W2 + b2
  gemm_k<2><<<dim3(kBT / 128, kC / 64), 256, 0, stream>>>(
      hbuf, W2T, kF, kC, out, ybuf, b2, nullptr, nullptr, nullptr, nullptr);
}
